// HierarchicalAttention_87711822119332
// MI455X (gfx1250) — matrix operand order repaired and checked by reading, not compiled, not run on hardware
//
#include <hip/hip_runtime.h>
#include <cstdint>
#include <cstddef>

typedef __bf16 bf16_t;
typedef bf16_t v16bf __attribute__((ext_vector_type(16)));
typedef float  v8f   __attribute__((ext_vector_type(8)));

#define DEV __device__ __forceinline__

DEV bf16_t f2bf(float f) {
  union { float f; unsigned u; } in; in.f = f;
  unsigned u = in.u + 0x7FFFu + ((in.u >> 16) & 1u);   // round-to-nearest-even
  union { unsigned short s; bf16_t b; } out; out.s = (unsigned short)(u >> 16);
  return out.b;
}

DEV v8f wmma_bf16(v16bf a, v16bf b, v8f c) {
  // D = A(16x32) * B(32x16) + C, fp32 accumulate
  return __builtin_amdgcn_wmma_f32_16x16x32_bf16(false, a, false, b, (short)0, c,
                                                 false, false);
}

union FragU { v16bf v; unsigned u[8]; };

// A fragment: 16x32 bf16 tile from row-major storage, leading dim ld (elements).
// ISA layout: lanes 0-15 row M=lane, VGPR v holds K-pair; lanes 16-31 K += 8.
DEV v16bf load_a_frag(const bf16_t* base, int ld) {
  const int lane = threadIdx.x & 31;
  const int g = lane >> 4, r = lane & 15;
  const bf16_t* p = base + (size_t)r * ld + g * 8;
  FragU f;
#pragma unroll
  for (int v = 0; v < 8; ++v) {
    int k = ((v & 4) ? 16 : 0) + (v & 3) * 2;
    f.u[v] = *(const unsigned*)(p + k);
  }
  return f.v;
}

// B fragment: 32x16 (KxN) bf16 from N-major storage Bt[n][k], leading dim ld.
// ISA layout: lane N=lane&15; VGPR v holds K-pair; lanes 16-31 K += 8 (same as A).
DEV v16bf load_b_frag(const bf16_t* base, int ld) {
  const int lane = threadIdx.x & 31;
  const int n = lane & 15, kb = (lane >> 4) * 8;
  const bf16_t* p = base + (size_t)n * ld + kb;
  FragU f;
#pragma unroll
  for (int v = 0; v < 8; ++v) f.u[v] = *(const unsigned*)(p + ((v & 4) ? 16 : 0) + (v & 3) * 2);
  return f.v;
}

// Async global->LDS copy of 32 bytes per lane (two B128 ops, ASYNCcnt).
// INST_OFFSET applies to both the LDS and global addresses (ISA 08 §4.4).
DEV void async_copy32(unsigned lds_off, const void* gptr) {
  uint64_t ga = (uint64_t)gptr;
  asm volatile(
      "global_load_async_to_lds_b128 %0, %1, off\n\t"
      "global_load_async_to_lds_b128 %0, %1, off offset:16"
      :: "v"(lds_off), "v"(ga)
      : "memory");
}

DEV unsigned lds_addr(const void* p) { return (unsigned)(size_t)p; }

// ---------------------------------------------------------------- converts --

__global__ void convert_bf16x4(const float* __restrict__ in,
                               bf16_t* __restrict__ out, long n4) {
  long i = (long)blockIdx.x * blockDim.x + threadIdx.x;
  if (i >= n4) return;
  float4 f = ((const float4*)in)[i];
  union { bf16_t b[4]; uint2 u; } o;
  o.b[0] = f2bf(f.x); o.b[1] = f2bf(f.y); o.b[2] = f2bf(f.z); o.b[3] = f2bf(f.w);
  ((uint2*)out)[i] = o.u;
}

// Wt[n][k] = bf16(W[k][n]); W is K x N fp32, K,N multiples of 32.
__global__ __launch_bounds__(256) void transpose_w_bf16(
    const float* __restrict__ W, bf16_t* __restrict__ Wt, int K, int N) {
  __shared__ float tile[32][33];
  const int tx = threadIdx.x & 31, ty = threadIdx.x >> 5;
  const int n0 = blockIdx.x * 32, k0 = blockIdx.y * 32;
#pragma unroll
  for (int i = ty; i < 32; i += 8)
    tile[i][tx] = W[(size_t)(k0 + i) * N + n0 + tx];
  __syncthreads();
#pragma unroll
  for (int i = ty; i < 32; i += 8)
    Wt[(size_t)(n0 + i) * K + k0 + tx] = f2bf(tile[tx][i]);
}

__global__ void gather_tokens(const bf16_t* __restrict__ xb,
                              bf16_t* __restrict__ xg,
                              int S, int D, int strideTok, int G, long total) {
  long i = (long)blockIdx.x * blockDim.x + threadIdx.x;
  if (i >= total) return;
  int d = (int)(i % D);
  long t = i / D;
  int gi = (int)(t % G);
  int b  = (int)(t / G);
  xg[i] = xb[((size_t)b * S + (size_t)gi * strideTok) * D + d];
}

// -------------------------------------------------------------------- GEMM --
// C(MxN) = A(MxK, bf16 row-major) * Wt(NxK, bf16 N-major)^T.
// Block tile 128x128x32, 256 threads = 8 waves, wave tile 32x64.
// Double-buffered LDS fed by async global->LDS loads (ASYNCcnt pipeline);
// the last k-step is peeled so the steady-state loop is branch-free.

constexpr int EPI_BF16   = 0;  // out bf16: (acc + bias) * scale
constexpr int EPI_VT     = 1;  // out bf16 into (B,H,DK,T) transposed layout
constexpr int EPI_F32RES = 2;  // out fp32: acc + bias + resid

template<int EPI>
__global__ __launch_bounds__(256) void gemm_bf16k(
    const bf16_t* __restrict__ A, int lda,
    const bf16_t* __restrict__ Wt, int ldw,
    const float* __restrict__ bias, float scale,
    void* __restrict__ Cout, int ldc,
    const float* __restrict__ resid,
    int K, int rowsPerBatch) {
  constexpr int LD = 40;                    // padded shorts/row: 80B, 16B-aligned
  __shared__ bf16_t As[2][128 * LD];
  __shared__ bf16_t Bs[2][128 * LD];
  const int tid = threadIdx.x, wid = tid >> 5, lane = tid & 31;
  const int m0 = blockIdx.y * 128, n0 = blockIdx.x * 128;
  const int wm = wid >> 1, wn = wid & 1;
  const int g = lane >> 4, ln = lane & 15;
  v8f acc[2][4];
#pragma unroll
  for (int mt = 0; mt < 2; ++mt)
#pragma unroll
    for (int nt = 0; nt < 4; ++nt)
#pragma unroll
      for (int v = 0; v < 8; ++v) acc[mt][nt][v] = 0.f;

  const int r = tid >> 1, cc = (tid & 1) * 16;
  const bf16_t* gA = A + (size_t)(m0 + r) * lda + cc;
  const bf16_t* gB = Wt + (size_t)(n0 + r) * ldw + cc;
  const unsigned laS = lds_addr(&As[0][r * LD + cc]);
  const unsigned lbS = lds_addr(&Bs[0][r * LD + cc]);
  const unsigned bufStride = (unsigned)(128 * LD * sizeof(bf16_t));

  auto issue = [&](int k0, int bufi) {
    async_copy32(laS + bufi * bufStride, gA + k0);
    async_copy32(lbS + bufi * bufStride, gB + k0);
  };
  auto compute = [&](int bufi) {
    v16bf a0 = load_a_frag(&As[bufi][(wm * 32) * LD], LD);
    v16bf a1 = load_a_frag(&As[bufi][(wm * 32 + 16) * LD], LD);
#pragma unroll
    for (int nt = 0; nt < 4; ++nt) {
      v16bf bb = load_b_frag(&Bs[bufi][(wn * 64 + nt * 16) * LD], LD);
      acc[0][nt] = wmma_bf16(a0, bb, acc[0][nt]);
      acc[1][nt] = wmma_bf16(a1, bb, acc[1][nt]);
    }
  };

  issue(0, 0);
  int buf = 0;
  for (int k0 = 0; k0 < K - 32; k0 += 32) {   // steady state: branch-free
    __syncthreads();                          // all waves done reading buf^1
    issue(k0 + 32, buf ^ 1);                  // 4 more in flight (8 total)
    asm volatile("s_wait_asynccnt 0x4" ::: "memory");  // tile k0 landed
    __syncthreads();                          // tile k0 visible to all waves
    compute(buf);
    buf ^= 1;
  }
  __syncthreads();                            // drain: final k-step
  asm volatile("s_wait_asynccnt 0x0" ::: "memory");
  __syncthreads();
  compute(buf);

#pragma unroll
  for (int mt = 0; mt < 2; ++mt)
#pragma unroll
    for (int nt = 0; nt < 4; ++nt)
#pragma unroll
      for (int v = 0; v < 8; ++v) {
        int row = m0 + wm * 32 + mt * 16 + v + 8 * g;
        int col = n0 + wn * 64 + nt * 16 + ln;
        float val = (acc[mt][nt][v] + bias[col]) * scale;
        if constexpr (EPI == EPI_BF16) {
          ((bf16_t*)Cout)[(size_t)row * ldc + col] = f2bf(val);
        } else if constexpr (EPI == EPI_VT) {
          int bb = row / rowsPerBatch, s = row - bb * rowsPerBatch;
          int hh = col >> 6, dk = col & 63;
          ((bf16_t*)Cout)[(((size_t)bb * 8 + hh) * 64 + dk) * rowsPerBatch + s] =
              f2bf(val);
        } else {
          ((float*)Cout)[(size_t)row * ldc + col] =
              val + resid[(size_t)row * ldc + col];
        }
      }
}

// --------------------------------------------------------------- attention --
// One WG per (query-chunk-of-256, head, batch). 8 waves, 32 query rows each.
// Q: (B,S,D) bf16 pre-scaled by 1/sqrt(DK). Kt: (B,RKV,D) token-major.
// Vt: (B,H,DK,RKV) feature-major (pre-transposed). Out: (B,S,2D) bf16 concat.

template<int NKV>
__global__ __launch_bounds__(256) void attn_kernel(
    const bf16_t* __restrict__ Q, const bf16_t* __restrict__ Kt,
    const bf16_t* __restrict__ Vt, bf16_t* __restrict__ Out,
    int S, int RKV, int outColOff, int kvLocal) {
  constexpr int D = 512, H = 8, DK = 64;
  constexpr int KS_LD = 72;  // padded shorts/row: 144B, 16B-aligned
  const int qc = blockIdx.x, h = blockIdx.y, b = blockIdx.z;
  const int tid = threadIdx.x, wid = tid >> 5, lane = tid & 31;
  const int g = lane >> 4, ln = lane & 15;

  extern __shared__ char smem[];
  bf16_t* Ks = (bf16_t*)smem;                        // [NKV][KS_LD]
  bf16_t* Ps = Ks + NKV * KS_LD + wid * (16 * 40);   // per-wave [16][40]

  const int kvBase = kvLocal ? qc * 256 : 0;
  {  // async cooperative K-tile load: NKV rows x 64 features
    const bf16_t* gk = Kt + ((size_t)b * RKV + kvBase) * D + h * DK;
#pragma unroll
    for (int e0 = 0; e0 < NKV * 4; e0 += 256) {
      int e = e0 + tid;
      int n = e >> 2, c = (e & 3) * 16;
      async_copy32(lds_addr(&Ks[n * KS_LD + c]), gk + (size_t)n * D + c);
    }
    asm volatile("s_wait_asynccnt 0x0" ::: "memory");
  }
  __syncthreads();

  const bf16_t* qbase = Q + ((size_t)b * S + qc * 256 + wid * 32) * D + h * DK;
  const bf16_t* vbase = Vt + (((size_t)b * H + h) * DK) * (size_t)RKV + kvBase;

#pragma unroll
  for (int mt = 0; mt < 2; ++mt) {
    v16bf aq0 = load_a_frag(qbase + mt * 16 * D, D);
    v16bf aq1 = load_a_frag(qbase + mt * 16 * D + 32, D);

    v8f sc[NKV / 16];
#pragma unroll
    for (int nt = 0; nt < NKV / 16; ++nt) {
#pragma unroll
      for (int v = 0; v < 8; ++v) sc[nt][v] = 0.f;
      v16bf b0 = load_b_frag(&Ks[nt * 16 * KS_LD], KS_LD);
      sc[nt] = wmma_bf16(aq0, b0, sc[nt]);
      v16bf b1 = load_b_frag(&Ks[nt * 16 * KS_LD + 32], KS_LD);
      sc[nt] = wmma_bf16(aq1, b1, sc[nt]);
    }

    // fp32 softmax; row M=v+8g lives across 16 lanes of one half-wave.
#pragma unroll
    for (int v = 0; v < 8; ++v) {
      float m = -3.0e38f;
#pragma unroll
      for (int nt = 0; nt < NKV / 16; ++nt) m = fmaxf(m, sc[nt][v]);
      m = fmaxf(m, __shfl_xor(m, 1));
      m = fmaxf(m, __shfl_xor(m, 2));
      m = fmaxf(m, __shfl_xor(m, 4));
      m = fmaxf(m, __shfl_xor(m, 8));
      float s = 0.f;
#pragma unroll
      for (int nt = 0; nt < NKV / 16; ++nt) {
        float e = __expf(sc[nt][v] - m);
        sc[nt][v] = e; s += e;
      }
      s += __shfl_xor(s, 1); s += __shfl_xor(s, 2);
      s += __shfl_xor(s, 4); s += __shfl_xor(s, 8);
      float inv = 1.f / s;
#pragma unroll
      for (int nt = 0; nt < NKV / 16; ++nt) sc[nt][v] *= inv;
    }

    // P @ V  (restage P through wave-private LDS to reach A-fragment layout)
    v8f ov[4];
#pragma unroll
    for (int dt = 0; dt < 4; ++dt)
#pragma unroll
      for (int v = 0; v < 8; ++v) ov[dt][v] = 0.f;
#pragma unroll
    for (int c = 0; c < NKV / 32; ++c) {
#pragma unroll
      for (int t = 0; t < 2; ++t) {
        int nt = c * 2 + t;
#pragma unroll
        for (int v = 0; v < 8; ++v)
          Ps[(v + 8 * g) * 40 + t * 16 + ln] = f2bf(sc[nt][v]);
      }
      asm volatile("s_wait_dscnt 0" ::: "memory");
      v16bf ap = load_a_frag(Ps, 40);
#pragma unroll
      for (int dt = 0; dt < 4; ++dt) {
        v16bf bv = load_b_frag(vbase + (size_t)(dt * 16) * RKV + c * 32, RKV);
        ov[dt] = wmma_bf16(ap, bv, ov[dt]);
      }
      asm volatile("" ::: "memory");
    }

    bf16_t* o = Out + ((size_t)b * S + qc * 256 + wid * 32 + mt * 16) * (2 * D) +
                outColOff + h * DK;
#pragma unroll
    for (int dt = 0; dt < 4; ++dt)
#pragma unroll
      for (int v = 0; v < 8; ++v)
        o[(size_t)(v + 8 * g) * (2 * D) + dt * 16 + ln] = f2bf(ov[dt][v]);
  }
}

// -------------------------------------------------------------- layer norm --

__global__ __launch_bounds__(256) void ln_kernel(
    float* __restrict__ hbuf, const float* __restrict__ gamma,
    const float* __restrict__ beta, size_t rows) {
  const int wid = threadIdx.x >> 5, lane = threadIdx.x & 31;
  size_t row = (size_t)blockIdx.x * 8 + wid;
  if (row >= rows) return;
  float* p = hbuf + row * 512;
  float4 vals[4];
  float sum = 0.f, sq = 0.f;
#pragma unroll
  for (int j = 0; j < 4; ++j) {
    vals[j] = ((float4*)p)[lane + j * 32];
    sum += vals[j].x + vals[j].y + vals[j].z + vals[j].w;
    sq += vals[j].x * vals[j].x + vals[j].y * vals[j].y +
          vals[j].z * vals[j].z + vals[j].w * vals[j].w;
  }
#pragma unroll
  for (int off = 16; off; off >>= 1) {
    sum += __shfl_xor(sum, off);
    sq  += __shfl_xor(sq, off);
  }
  float mean = sum * (1.f / 512.f);
  float var  = sq * (1.f / 512.f) - mean * mean;
  float inv  = rsqrtf(var + 1e-5f);
#pragma unroll
  for (int j = 0; j < 4; ++j) {
    int c0 = (lane + j * 32) * 4;
    float4 v = vals[j];
    v.x = (v.x - mean) * inv * gamma[c0]     + beta[c0];
    v.y = (v.y - mean) * inv * gamma[c0 + 1] + beta[c0 + 1];
    v.z = (v.z - mean) * inv * gamma[c0 + 2] + beta[c0 + 2];
    v.w = (v.w - mean) * inv * gamma[c0 + 3] + beta[c0 + 3];
    ((float4*)p)[lane + j * 32] = v;
  }
}

// ------------------------------------------------------------------ launch --

extern "C" void kernel_launch(void* const* d_in, const int* in_sizes, int n_in,
                              void* d_out, int out_size, void* d_ws,
                              size_t ws_size, hipStream_t stream) {
  (void)in_sizes; (void)n_in; (void)out_size; (void)ws_size;
  constexpr int B = 4, S = 8192, D = 512, H = 8, G = 128;
  const float* x     = (const float*)d_in[0];
  const float* lq_w  = (const float*)d_in[1];
  const float* lq_b  = (const float*)d_in[2];
  const float* lk_w  = (const float*)d_in[3];
  const float* lk_b  = (const float*)d_in[4];
  const float* lv_w  = (const float*)d_in[5];
  const float* lv_b  = (const float*)d_in[6];
  const float* gq_w  = (const float*)d_in[7];
  const float* gq_b  = (const float*)d_in[8];
  const float* gk_w  = (const float*)d_in[9];
  const float* gk_b  = (const float*)d_in[10];
  const float* gv_w  = (const float*)d_in[11];
  const float* gv_b  = (const float*)d_in[12];
  const float* out_w = (const float*)d_in[13];
  const float* out_b = (const float*)d_in[14];
  const float* ln_g  = (const float*)d_in[15];
  const float* ln_b  = (const float*)d_in[16];

  char* ws = (char*)d_ws;
  size_t off = 0;
  auto alloc = [&](size_t bytes) -> char* {
    char* p = ws + off;
    off += (bytes + 255) & ~(size_t)255;
    return p;
  };
  bf16_t* xb   = (bf16_t*)alloc((size_t)B * S * D * 2);
  bf16_t* xg   = (bf16_t*)alloc((size_t)B * G * D * 2);
  bf16_t* wlq  = (bf16_t*)alloc((size_t)D * D * 2);
  bf16_t* wlk  = (bf16_t*)alloc((size_t)D * D * 2);
  bf16_t* wlv  = (bf16_t*)alloc((size_t)D * D * 2);
  bf16_t* wgq  = (bf16_t*)alloc((size_t)D * D * 2);
  bf16_t* wgk  = (bf16_t*)alloc((size_t)D * D * 2);
  bf16_t* wgv  = (bf16_t*)alloc((size_t)D * D * 2);
  bf16_t* wout = (bf16_t*)alloc((size_t)D * 2 * D * 2);
  bf16_t* ql   = (bf16_t*)alloc((size_t)B * S * D * 2);
  bf16_t* kl   = (bf16_t*)alloc((size_t)B * S * D * 2);
  bf16_t* vlT  = (bf16_t*)alloc((size_t)B * S * D * 2);
  bf16_t* qg   = (bf16_t*)alloc((size_t)B * S * D * 2);
  bf16_t* kg   = (bf16_t*)alloc((size_t)B * G * D * 2);
  bf16_t* vgT  = (bf16_t*)alloc((size_t)B * G * D * 2);
  bf16_t* attn = (bf16_t*)alloc((size_t)B * S * 2 * D * 2);

  // 1) precision conversion + weight transposes + global-token gather
  {
    long n4 = (long)B * S * D / 4;
    convert_bf16x4<<<(unsigned)((n4 + 255) / 256), 256, 0, stream>>>(x, xb, n4);
  }
  transpose_w_bf16<<<dim3(D / 32, D / 32), 256, 0, stream>>>(lq_w, wlq, D, D);
  transpose_w_bf16<<<dim3(D / 32, D / 32), 256, 0, stream>>>(lk_w, wlk, D, D);
  transpose_w_bf16<<<dim3(D / 32, D / 32), 256, 0, stream>>>(lv_w, wlv, D, D);
  transpose_w_bf16<<<dim3(D / 32, D / 32), 256, 0, stream>>>(gq_w, wgq, D, D);
  transpose_w_bf16<<<dim3(D / 32, D / 32), 256, 0, stream>>>(gk_w, wgk, D, D);
  transpose_w_bf16<<<dim3(D / 32, D / 32), 256, 0, stream>>>(gv_w, wgv, D, D);
  transpose_w_bf16<<<dim3(D / 32, (2 * D) / 32), 256, 0, stream>>>(out_w, wout,
                                                                  2 * D, D);
  {
    long tot = (long)B * G * D;
    gather_tokens<<<(unsigned)((tot + 255) / 256), 256, 0, stream>>>(
        xb, xg, S, D, 64, G, tot);
  }

  // 2) projections (scale 1/sqrt(64)=0.125 folded into q)
  dim3 gBig(D / 128, (B * S) / 128);
  dim3 gSmall(D / 128, (B * G) / 128);
  gemm_bf16k<EPI_BF16><<<gBig, 256, 0, stream>>>(xb, D, wlq, D, lq_b, 0.125f,
                                                 ql, D, nullptr, D, 1);
  gemm_bf16k<EPI_BF16><<<gBig, 256, 0, stream>>>(xb, D, wlk, D, lk_b, 1.0f,
                                                 kl, D, nullptr, D, 1);
  gemm_bf16k<EPI_VT><<<gBig, 256, 0, stream>>>(xb, D, wlv, D, lv_b, 1.0f,
                                               vlT, D, nullptr, D, S);
  gemm_bf16k<EPI_BF16><<<gBig, 256, 0, stream>>>(xb, D, wgq, D, gq_b, 0.125f,
                                                 qg, D, nullptr, D, 1);
  gemm_bf16k<EPI_BF16><<<gSmall, 256, 0, stream>>>(xg, D, wgk, D, gk_b, 1.0f,
                                                   kg, D, nullptr, D, 1);
  gemm_bf16k<EPI_VT><<<gSmall, 256, 0, stream>>>(xg, D, wgv, D, gv_b, 1.0f,
                                                 vgT, D, nullptr, D, G);

  // 3) attention → concat buffer (local: cols [0,512), global: cols [512,1024))
  size_t smemL = ((size_t)256 * 72 + 8 * 16 * 40) * sizeof(bf16_t);
  size_t smemG = ((size_t)128 * 72 + 8 * 16 * 40) * sizeof(bf16_t);
  attn_kernel<256><<<dim3(S / 256, H, B), 256, smemL, stream>>>(
      ql, kl, vlT, attn, S, S, 0, 1);
  attn_kernel<128><<<dim3(S / 256, H, B), 256, smemG, stream>>>(
      qg, kg, vgT, attn, S, G, D, 0);

  // 4) output projection + residual, then in-place LayerNorm on d_out
  gemm_bf16k<EPI_F32RES><<<gBig, 256, 0, stream>>>(
      attn, 2 * D, wout, 2 * D, out_b, 1.0f, (float*)d_out, D, x, 2 * D, 1);
  ln_kernel<<<(B * S) / 8, 256, 0, stream>>>((float*)d_out, ln_g, ln_b,
                                             (size_t)B * S);
}
